// SSA_SC_30880814858731
// MI455X (gfx1250) — matrix operand order repaired and checked by reading, not compiled, not run on hardware
//
#include <hip/hip_runtime.h>
#include <hip/hip_bf16.h>

typedef __attribute__((ext_vector_type(16))) __bf16 bf16x16;
typedef __attribute__((ext_vector_type(8)))  float  floatx8;

#define EPI_STATS   0
#define EPI_SCATTER 1
#define EPI_BEV     2
#define EPI_VOX     3

#define GRID_B 4
#define GRID_X 256
#define GRID_Z 256
#define C_CMP  32
#define GRID_ELEMS (GRID_B * C_CMP * GRID_X * GRID_Z)  // 8388608

#define NWAVES 4  // waves per block in WMMA kernels

__device__ __forceinline__ void atomicMaxF(float* addr, float val) {
    // valid when buffer is initialized to -inf
    if (val >= 0.0f) atomicMax((int*)addr, __float_as_int(val));
    else             atomicMin((unsigned int*)addr, __float_as_uint(val));
}

// ---------------------------------------------------------------------------
// init: zero stats region, -inf pooled/vox, zero temp grid
// ---------------------------------------------------------------------------
__global__ void init_ws_kernel(float* stats, long long nstats,
                               float* pooled, long long npooled,
                               float* voxb, long long nvox,
                               float* tgrid, long long ngrid) {
    long long stride = (long long)gridDim.x * blockDim.x;
    long long i0 = (long long)blockIdx.x * blockDim.x + threadIdx.x;
    for (long long i = i0; i < nstats; i += stride) stats[i] = 0.0f;
    for (long long i = i0; i < npooled; i += stride) pooled[i] = -__builtin_inff();
    for (long long i = i0; i < nvox; i += stride) voxb[i] = -__builtin_inff();
    for (long long i = i0; i < ngrid; i += stride) tgrid[i] = 0.0f;
}

// ---------------------------------------------------------------------------
// input (3-channel) batch stats
// ---------------------------------------------------------------------------
__global__ void input_stats_kernel(const float* __restrict__ pt, int N,
                                   float* __restrict__ gsum, float* __restrict__ gsq) {
    __shared__ float sS[3], sQ[3];
    int tid = threadIdx.x;
    if (tid < 3) { sS[tid] = 0.0f; sQ[tid] = 0.0f; }
    __syncthreads();
    float s0 = 0, s1 = 0, s2 = 0, q0 = 0, q1 = 0, q2 = 0;
    for (int n = blockIdx.x * blockDim.x + tid; n < N; n += gridDim.x * blockDim.x) {
        float a = pt[(size_t)n * 3 + 0];
        float b = pt[(size_t)n * 3 + 1];
        float c = pt[(size_t)n * 3 + 2];
        s0 += a; q0 += a * a; s1 += b; q1 += b * b; s2 += c; q2 += c * c;
    }
    atomicAdd(&sS[0], s0); atomicAdd(&sQ[0], q0);
    atomicAdd(&sS[1], s1); atomicAdd(&sQ[1], q1);
    atomicAdd(&sS[2], s2); atomicAdd(&sQ[2], q2);
    __syncthreads();
    if (tid < 3) { atomicAdd(&gsum[tid], sS[tid]); atomicAdd(&gsq[tid], sQ[tid]); }
}

// ---------------------------------------------------------------------------
// fold BN stats into per-channel scale/shift: y = x*scale + shift
// ---------------------------------------------------------------------------
__global__ void finalize_bn_kernel(const float* __restrict__ gsum, const float* __restrict__ gsq,
                                   const float* __restrict__ g, const float* __restrict__ be,
                                   float* __restrict__ scale, float* __restrict__ shift,
                                   int C, float invN) {
    int c = blockIdx.x * blockDim.x + threadIdx.x;
    if (c < C) {
        float m = gsum[c] * invN;
        float v = gsq[c] * invN - m * m;
        float s = g[c] * rsqrtf(v + 1e-5f);
        scale[c] = s;
        shift[c] = be[c] - m * s;
    }
}

// ---------------------------------------------------------------------------
// layer 1: 3 -> 64, scalar (K=3 too small for WMMA), bf16 out + column stats
// ---------------------------------------------------------------------------
__global__ __launch_bounds__(256) void layer1_kernel(
        const float* __restrict__ pt, int N,
        const float* __restrict__ sc0, const float* __restrict__ sh0,
        const float* __restrict__ w1, const float* __restrict__ b1,
        __bf16* __restrict__ t1, float* __restrict__ gsum, float* __restrict__ gsq) {
    __shared__ float sSum[64], sSq[64], sw[192], sb[64], ssc[3], ssh[3];
    int tid = threadIdx.x;
    if (tid < 64) { sSum[tid] = 0.0f; sSq[tid] = 0.0f; sb[tid] = b1[tid]; }
    for (int i = tid; i < 192; i += blockDim.x) sw[i] = w1[i];
    if (tid < 3) { ssc[tid] = sc0[tid]; ssh[tid] = sh0[tid]; }
    __syncthreads();
    for (int n = blockIdx.x * blockDim.x + tid; n < N; n += gridDim.x * blockDim.x) {
        float x0 = pt[(size_t)n * 3 + 0] * ssc[0] + ssh[0];
        float x1 = pt[(size_t)n * 3 + 1] * ssc[1] + ssh[1];
        float x2 = pt[(size_t)n * 3 + 2] * ssc[2] + ssh[2];
        __bf16* orow = t1 + (size_t)n * 64;
        #pragma unroll 8
        for (int j = 0; j < 64; ++j) {
            float t = sb[j] + x0 * sw[j] + x1 * sw[64 + j] + x2 * sw[128 + j];
            orow[j] = (__bf16)t;
            atomicAdd(&sSum[j], t);
            atomicAdd(&sSq[j], t * t);
        }
    }
    __syncthreads();
    if (tid < 64) { atomicAdd(&gsum[tid], sSum[tid]); atomicAdd(&gsq[tid], sSq[tid]); }
}

// ---------------------------------------------------------------------------
// WMMA n-tile sweep + epilogue for one 16-row tile (FULL = no bounds checks)
// ---------------------------------------------------------------------------
template<int CIN, int COUT, int EPI, bool FULL>
__device__ __forceinline__ void compute_tiles_nt(
        const bf16x16 (&afrag)[CIN / 32],
        const __bf16* __restrict__ sW,
        const float* __restrict__ sBias,
        float* __restrict__ sSum, float* __restrict__ sSq,
        int rowBase, int M, int lane,
        __bf16* __restrict__ outbf, float* __restrict__ outf,
        const int* __restrict__ invA, const int* __restrict__ invB,
        float* __restrict__ pooled, float* __restrict__ voxb,
        const int* __restrict__ unq, float* __restrict__ grid_out) {
    const int col16  = lane & 15;
    const int khalfB = (lane >= 16) ? 16 : 0;   // B-matrix K half offset
    const int rAdd   = (lane >= 16) ? 8 : 0;    // C/D row offset
    const int rowTop = rowBase + rAdd;          // first of this lane's 8 output rows

    #pragma unroll
    for (int nt = 0; nt < COUT / 16; ++nt) {
        const int nBase = nt << 4;
        floatx8 acc = {};
        #pragma unroll
        for (int kt = 0; kt < CIN / 32; ++kt) {
            // B layout (32x16 bf16): N = lane%16, K = khalfB + 2v + t; 32B vector load
            const bf16x16 b = *reinterpret_cast<const bf16x16*>(
                    &sW[(nBase + col16) * CIN + kt * 32 + khalfB]);
            acc = __builtin_amdgcn_wmma_f32_16x16x32_bf16(
                    false, afrag[kt], false, b, (short)0, acc, false, false);
        }

        const int colOut = nBase + col16;        // C/D layout: N = lane%16
        const float bv = sBias[colOut];
        if constexpr (EPI == EPI_STATS) {
            __bf16* orow = outbf + (size_t)rowTop * COUT + colOut;
            float ls = 0.0f, lq = 0.0f;
            #pragma unroll
            for (int i = 0; i < 8; ++i) {
                if (FULL || rowTop + i < M) {
                    float vv = acc[i] + bv;
                    orow[i * COUT] = (__bf16)vv;   // constant imm offsets
                    ls += vv; lq += vv * vv;
                }
            }
            atomicAdd(&sSum[colOut], ls);
            atomicAdd(&sSq[colOut], lq);
        } else if constexpr (EPI == EPI_SCATTER) {
            const int* ia = invA + rowTop;
            const int* ib = invB + rowTop;
            #pragma unroll
            for (int i = 0; i < 8; ++i) {
                if (FULL || rowTop + i < M) {
                    float vv = acc[i] + bv;
                    atomicMaxF(&pooled[(size_t)ia[i] * COUT + colOut], vv);
                    atomicMaxF(&voxb[(size_t)ib[i] * COUT + colOut], vv);
                }
            }
        } else if constexpr (EPI == EPI_BEV) {
            const int* uq = unq + (size_t)rowTop * 3;
            #pragma unroll
            for (int i = 0; i < 8; ++i) {
                if (FULL || rowTop + i < M) {
                    float vv = fmaxf(acc[i] + bv, 0.0f);
                    const int bb = uq[i * 3 + 0];
                    const int xx = uq[i * 3 + 1];
                    const int zz = uq[i * 3 + 2];
                    grid_out[(((size_t)bb * GRID_X + xx) * GRID_Z + zz) * C_CMP + colOut] = vv;
                }
            }
        } else { // EPI_VOX
            float* orow = outf + (size_t)rowTop * COUT + colOut;
            #pragma unroll
            for (int i = 0; i < 8; ++i) {
                if (FULL || rowTop + i < M) {
                    orow[i * COUT] = fmaxf(acc[i] + bv, 0.0f);
                }
            }
        }
    }
}

// ---------------------------------------------------------------------------
// Fused WMMA GEMM layer: epi( relu_in(in*scale+shift) @ W + bias )
//   A staged global->LDS with global_load_async_to_lds_b128 (ASYNCcnt),
//   normalized+bf16-converted on fragment build; W cached bf16 (transposed)
//   in LDS; v_wmma_f32_16x16x32_bf16, fp32 accumulate.
// ---------------------------------------------------------------------------
template<int CIN, int COUT, int EPI, bool INBF, bool RELU>
__global__ __launch_bounds__(NWAVES * 32) void wmma_layer_kernel(
        const void* __restrict__ in, int M,
        const float* __restrict__ scale, const float* __restrict__ shift,
        const float* __restrict__ W, const float* __restrict__ bias,
        __bf16* __restrict__ outbf, float* __restrict__ outf,
        float* __restrict__ gsum, float* __restrict__ gsq,
        const int* __restrict__ invA, const int* __restrict__ invB,
        float* __restrict__ pooled, float* __restrict__ voxb,
        const int* __restrict__ unq, float* __restrict__ grid_out) {
    constexpr int ESZ = INBF ? 2 : 4;
    constexpr int TILE_BYTES = 16 * CIN * ESZ;

    __shared__ __bf16 sW[CIN * COUT];                       // transposed: sW[n*CIN+k]
    __shared__ float sScale[CIN], sShift[CIN], sBias[COUT];
    __shared__ float sSum[(EPI == EPI_STATS) ? COUT : 1];
    __shared__ float sSq[(EPI == EPI_STATS) ? COUT : 1];
    __shared__ alignas(16) unsigned char sA[NWAVES * TILE_BYTES];  // per-wave A staging

    const int tid  = threadIdx.x;
    const int lane = tid & 31;
    const int wave = tid >> 5;

    for (int i = tid; i < CIN * COUT; i += blockDim.x) {
        int n = i / CIN, k = i - n * CIN;
        sW[i] = (__bf16)W[k * COUT + n];
    }
    for (int i = tid; i < CIN; i += blockDim.x) {
        sScale[i] = scale ? scale[i] : 1.0f;
        sShift[i] = shift ? shift[i] : 0.0f;
    }
    for (int i = tid; i < COUT; i += blockDim.x) sBias[i] = bias[i];
    if constexpr (EPI == EPI_STATS) {
        for (int i = tid; i < COUT; i += blockDim.x) { sSum[i] = 0.0f; sSq[i] = 0.0f; }
    }
    __syncthreads();

    unsigned char* waveBuf = sA + (size_t)wave * TILE_BYTES;
    const unsigned ldsBase = (unsigned)(size_t)waveBuf;     // low 32 bits = LDS offset
    const int m16 = lane & 15;
    const int khalfA = (lane >= 16) ? 16 : 0;               // A-matrix K half offset
    const int mTiles = (M + 15) >> 4;

    for (int tile = blockIdx.x * NWAVES + wave; tile < mTiles; tile += gridDim.x * NWAVES) {
        const int rowBase = tile << 4;

        if (rowBase + 16 <= M) {
            // ---- full tile: async-stage contiguous 16xCIN block into LDS ----
            const unsigned char* gbase = (const unsigned char*)in + (size_t)rowBase * CIN * ESZ;
            #pragma unroll
            for (int off = lane * 16; off < TILE_BYTES; off += 32 * 16) {
                unsigned la = ldsBase + off;
                unsigned long long ga = (unsigned long long)(gbase + off);
                asm volatile("global_load_async_to_lds_b128 %0, %1, off"
                             :: "v"(la), "v"(ga) : "memory");
            }
            asm volatile("s_wait_asynccnt 0x0" ::: "memory");

            bf16x16 afrag[CIN / 32];
            #pragma unroll
            for (int kt = 0; kt < CIN / 32; ++kt) {
                bf16x16 a;
                #pragma unroll
                for (int v = 0; v < 8; ++v) {
                    const int kk = 2 * v + khalfA;
                    const int k = kt * 32 + kk;
                    float x0, x1;
                    if constexpr (INBF) {
                        const __bf16* aRow = (const __bf16*)waveBuf + (size_t)m16 * CIN;
                        x0 = (float)aRow[k]; x1 = (float)aRow[k + 1];
                    } else {
                        const float* aRow = (const float*)waveBuf + (size_t)m16 * CIN;
                        x0 = aRow[k]; x1 = aRow[k + 1];
                    }
                    x0 = x0 * sScale[k] + sShift[k];
                    x1 = x1 * sScale[k + 1] + sShift[k + 1];
                    if constexpr (RELU) { x0 = fmaxf(x0, 0.0f); x1 = fmaxf(x1, 0.0f); }
                    a[2 * v]     = (__bf16)x0;
                    a[2 * v + 1] = (__bf16)x1;
                }
                afrag[kt] = a;
            }
            compute_tiles_nt<CIN, COUT, EPI, true>(afrag, sW, sBias, sSum, sSq,
                    rowBase, M, lane, outbf, outf, invA, invB, pooled, voxb, unq, grid_out);
        } else {
            // ---- tail tile: guarded direct global loads ----
            const int rowA = rowBase + m16;
            const bool rowValid = (rowA < M);
            bf16x16 afrag[CIN / 32];
            #pragma unroll
            for (int kt = 0; kt < CIN / 32; ++kt) {
                bf16x16 a;
                #pragma unroll
                for (int v = 0; v < 8; ++v) {
                    const int kk = 2 * v + khalfA;
                    const int k = kt * 32 + kk;
                    float x0 = 0.0f, x1 = 0.0f;
                    if (rowValid) {
                        if constexpr (INBF) {
                            const __bf16* inb = (const __bf16*)in;
                            x0 = (float)inb[(size_t)rowA * CIN + k];
                            x1 = (float)inb[(size_t)rowA * CIN + k + 1];
                        } else {
                            const float* inf = (const float*)in;
                            x0 = inf[(size_t)rowA * CIN + k];
                            x1 = inf[(size_t)rowA * CIN + k + 1];
                        }
                        x0 = x0 * sScale[k] + sShift[k];
                        x1 = x1 * sScale[k + 1] + sShift[k + 1];
                        if constexpr (RELU) { x0 = fmaxf(x0, 0.0f); x1 = fmaxf(x1, 0.0f); }
                    }
                    a[2 * v]     = (__bf16)x0;
                    a[2 * v + 1] = (__bf16)x1;
                }
                afrag[kt] = a;
            }
            compute_tiles_nt<CIN, COUT, EPI, false>(afrag, sW, sBias, sSum, sSq,
                    rowBase, M, lane, outbf, outf, invA, invB, pooled, voxb, unq, grid_out);
        }
    }

    if constexpr (EPI == EPI_STATS) {
        __syncthreads();
        for (int i = tid; i < COUT; i += blockDim.x) {
            atomicAdd(&gsum[i], sSum[i]);
            atomicAdd(&gsq[i], sSq[i]);
        }
    }
}

// ---------------------------------------------------------------------------
// 3x3 stride-1 pad-1 maxpool over [B,X,Z,C] temp grid -> out [B,C,X,Z]
// ---------------------------------------------------------------------------
__global__ __launch_bounds__(256) void maxpool_kernel(const float* __restrict__ g,
                                                      float* __restrict__ out) {
    int idx = blockIdx.x * blockDim.x + threadIdx.x;
    if (idx >= GRID_ELEMS) return;
    int c = idx & (C_CMP - 1);
    int z = (idx >> 5) & (GRID_Z - 1);
    int x = (idx >> 13) & (GRID_X - 1);
    int b = idx >> 21;
    float m = -__builtin_inff();
    #pragma unroll
    for (int dx = -1; dx <= 1; ++dx) {
        int xx = x + dx;
        if (xx < 0 || xx >= GRID_X) continue;
        #pragma unroll
        for (int dz = -1; dz <= 1; ++dz) {
            int zz = z + dz;
            if (zz < 0 || zz >= GRID_Z) continue;
            m = fmaxf(m, g[(((size_t)b * GRID_X + xx) * GRID_Z + zz) * C_CMP + c]);
        }
    }
    out[(((size_t)b * C_CMP + c) * GRID_X + x) * GRID_Z + z] = m;
}

// ---------------------------------------------------------------------------
extern "C" void kernel_launch(void* const* d_in, const int* in_sizes, int n_in,
                              void* d_out, int out_size, void* d_ws, size_t ws_size,
                              hipStream_t stream) {
    const float* pt   = (const float*)d_in[0];
    const float* bn0g = (const float*)d_in[1];
    const float* bn0b = (const float*)d_in[2];
    const float* w1   = (const float*)d_in[3];
    const float* b1   = (const float*)d_in[4];
    const float* g1   = (const float*)d_in[5];
    const float* be1  = (const float*)d_in[6];
    const float* w2   = (const float*)d_in[7];
    const float* b2   = (const float*)d_in[8];
    const float* g2   = (const float*)d_in[9];
    const float* be2  = (const float*)d_in[10];
    const float* w3   = (const float*)d_in[11];
    const float* b3   = (const float*)d_in[12];
    const float* g3   = (const float*)d_in[13];
    const float* be3  = (const float*)d_in[14];
    const float* w4   = (const float*)d_in[15];
    const float* b4   = (const float*)d_in[16];
    const float* wc   = (const float*)d_in[17];
    const float* bc   = (const float*)d_in[18];
    const float* wv   = (const float*)d_in[19];
    const float* bv   = (const float*)d_in[20];
    const int* unq_inv     = (const int*)d_in[21];
    const int* unq         = (const int*)d_in[22];
    const int* unq_inv_xyz = (const int*)d_in[24];

    const int N  = in_sizes[0] / 3;
    const int U  = in_sizes[22] / 3;
    const int Uv = in_sizes[23] / 4;
    const float invN = 1.0f / (float)N;

    // workspace layout
    float* wsf  = (float*)d_ws;                 // stats + folded scale/shift (8 KB)
    float* sum0 = wsf + 0;    float* sq0 = wsf + 8;     // 3
    float* sum1 = wsf + 16;   float* sq1 = wsf + 80;    // 64
    float* sum2 = wsf + 144;  float* sq2 = wsf + 272;   // 128
    float* sum3 = wsf + 400;  float* sq3 = wsf + 656;   // 256
    float* sc0  = wsf + 1024; float* sh0 = wsf + 1032;  // 3
    float* sc1  = wsf + 1040; float* sh1 = wsf + 1104;  // 64
    float* sc2  = wsf + 1168; float* sh2 = wsf + 1296;  // 128
    float* sc3  = wsf + 1424; float* sh3 = wsf + 1680;  // 256
    char* wsb = (char*)d_ws;
    __bf16* t1 = (__bf16*)(wsb + 8192);                  // [N,64]  bf16
    __bf16* t2 = t1 + (size_t)N * 64;                    // [N,128] bf16
    __bf16* t3 = t2 + (size_t)N * 128;                   // [N,256] bf16
    float* pooledBuf = (float*)(t3 + (size_t)N * 256);   // [U,128]  f32
    float* voxBuf    = pooledBuf + (size_t)U * 128;      // [Uv,128] f32
    float* tgrid     = voxBuf + (size_t)Uv * 128;        // [B,X,Z,32] f32

    float* outGrid = (float*)d_out;
    float* outVox  = (float*)d_out + (size_t)GRID_ELEMS;

    // 1) init scratch
    init_ws_kernel<<<2048, 256, 0, stream>>>(
        wsf, 1024, pooledBuf, (long long)U * 128,
        voxBuf, (long long)Uv * 128, tgrid, (long long)GRID_ELEMS);

    // 2) input BN stats + fold
    input_stats_kernel<<<1024, 256, 0, stream>>>(pt, N, sum0, sq0);
    finalize_bn_kernel<<<1, 64, 0, stream>>>(sum0, sq0, bn0g, bn0b, sc0, sh0, 3, invN);

    // 3) layer 1: 3->64 (scalar) + stats
    layer1_kernel<<<2048, 256, 0, stream>>>(pt, N, sc0, sh0, w1, b1, t1, sum1, sq1);
    finalize_bn_kernel<<<1, 64, 0, stream>>>(sum1, sq1, g1, be1, sc1, sh1, 64, invN);

    const int mtN = (N + 15) / 16;
    const int blkN = (mtN + NWAVES - 1) / NWAVES;

    // 4) layer 2: 64->128 (WMMA bf16) + stats
    wmma_layer_kernel<64, 128, EPI_STATS, true, true><<<blkN, NWAVES * 32, 0, stream>>>(
        t1, N, sc1, sh1, w2, b2, t2, nullptr, sum2, sq2,
        nullptr, nullptr, nullptr, nullptr, nullptr, nullptr);
    finalize_bn_kernel<<<1, 128, 0, stream>>>(sum2, sq2, g2, be2, sc2, sh2, 128, invN);

    // 5) layer 3: 128->256 (WMMA bf16) + stats
    wmma_layer_kernel<128, 256, EPI_STATS, true, true><<<blkN, NWAVES * 32, 0, stream>>>(
        t2, N, sc2, sh2, w3, b3, t3, nullptr, sum3, sq3,
        nullptr, nullptr, nullptr, nullptr, nullptr, nullptr);
    finalize_bn_kernel<<<1, 256, 0, stream>>>(sum3, sq3, g3, be3, sc3, sh3, 256, invN);

    // 6) layer 4: 256->128 (WMMA bf16), fused segment-max scatter (h never stored)
    wmma_layer_kernel<256, 128, EPI_SCATTER, true, true><<<blkN, NWAVES * 32, 0, stream>>>(
        t3, N, sc3, sh3, w4, b4, nullptr, nullptr, nullptr, nullptr,
        unq_inv, unq_inv_xyz, pooledBuf, voxBuf, nullptr, nullptr);

    // 7) BEV compression 128->32 + ReLU -> dense [B,X,Z,32] grid
    const int mtU = (U + 15) / 16;
    wmma_layer_kernel<128, 32, EPI_BEV, false, false><<<(mtU + NWAVES - 1) / NWAVES, NWAVES * 32, 0, stream>>>(
        pooledBuf, U, nullptr, nullptr, wc, bc, nullptr, nullptr, nullptr, nullptr,
        nullptr, nullptr, nullptr, nullptr, unq, tgrid);

    // 8) voxel compression 128->32 + ReLU -> d_out tail
    const int mtV = (Uv + 15) / 16;
    wmma_layer_kernel<128, 32, EPI_VOX, false, false><<<(mtV + NWAVES - 1) / NWAVES, NWAVES * 32, 0, stream>>>(
        voxBuf, Uv, nullptr, nullptr, wv, bv, nullptr, outVox, nullptr, nullptr,
        nullptr, nullptr, nullptr, nullptr, nullptr, nullptr);

    // 9) 3x3 maxpool + transpose -> d_out grid [B,C,X,Z]
    maxpool_kernel<<<(GRID_ELEMS + 255) / 256, 256, 0, stream>>>(tgrid, outGrid);
}
